// GCL_10557029613792
// MI455X (gfx1250) — matrix operand order repaired and checked by reading, not compiled, not run on hardware
//
#include <hip/hip_runtime.h>
#include <hip/hip_fp16.h>

typedef __attribute__((ext_vector_type(16))) _Float16 v16h;
typedef __attribute__((ext_vector_type(8)))  _Float16 v8h;
typedef __attribute__((ext_vector_type(8)))  float    v8f;
typedef __attribute__((ext_vector_type(4)))  float    v4f;

#define LDSTRIDE 136   // 128 + 8 f16 pad -> 272B row stride, kills LDS bank conflicts

// D = A(16xK f16) * B(Kx16 f16) + C(16x16 f32), wave32
static __device__ __forceinline__ v8f wmma16(v16h a, v16h b, v8f c) {
  return __builtin_amdgcn_wmma_f32_16x16x32_f16(false, a, false, b, (short)0, c, false, false);
}

// Build one 16x32 f16 A-fragment from a row-major f32 source.
// Layout (ISA 7.12.2): lane (r = lane&15, half = lane>>4) holds row r;
// v16h elems 0..7  = K = half*8 + 0..7
// v16h elems 8..15 = K = 16 + half*8 + 0..7
static __device__ __forceinline__ v16h a_frag_f32(const float* base, int half) {
  v4f x0 = *(const v4f*)(base + half * 8);
  v4f x1 = *(const v4f*)(base + half * 8 + 4);
  v4f x2 = *(const v4f*)(base + 16 + half * 8);
  v4f x3 = *(const v4f*)(base + 16 + half * 8 + 4);
  v16h a;
#pragma unroll
  for (int i = 0; i < 4; ++i) {
    a[i]      = (_Float16)x0[i];
    a[4 + i]  = (_Float16)x1[i];
    a[8 + i]  = (_Float16)x2[i];
    a[12 + i] = (_Float16)x3[i];
  }
  return a;
}

// B-fragment (32x16) from transposed f16 weights WT[n][k] (row n = output col).
// Lane (c = lane&15, half): elems 0..7 = K half*8+0..7, elems 8..15 = K 16+half*8+0..7 (WT[c][K])
static __device__ __forceinline__ v16h b_frag(const _Float16* __restrict__ WT,
                                              int ldk, int nt, int kf, int c, int half) {
  const _Float16* p = WT + (size_t)(nt * 16 + c) * ldk + kf * 32 + half * 8;
  v16h b;
  *(v8h*)&b       = *(const v8h*)p;
  *((v8h*)&b + 1) = *(const v8h*)(p + 16);
  return b;
}

// A-fragment from an f16 LDS row (same layout rule as a_frag_f32)
static __device__ __forceinline__ v16h a_frag_lds(const _Float16* rowp, int kf, int half) {
  const _Float16* p0 = rowp + kf * 32 + half * 8;
  v16h a;
  *(v8h*)&a       = *(const v8h*)p0;
  *((v8h*)&a + 1) = *(const v8h*)(p0 + 16);
  return a;
}

// ---------------- edge model ----------------
// per wave: 16 edges. X=[16,192] -> W1[192,128] -> +b, LN, SiLU -> W2[128,64] -> +b2+ef
// then e_out store + atomic scatter-add into agg[row].
__global__ __launch_bounds__(256) void edge_kernel(
    const float* __restrict__ h, const float* __restrict__ ef,
    const _Float16* __restrict__ W1T, const float* __restrict__ b1,
    const float* __restrict__ g, const float* __restrict__ bt,
    const _Float16* __restrict__ W2T, const float* __restrict__ b2,
    const int* __restrict__ erow, const int* __restrict__ ecol,
    float* __restrict__ e_out, float* __restrict__ agg,
    int E, int n_tiles) {
  __shared__ _Float16 Y[8][16][LDSTRIDE];
  const int wave = threadIdx.x >> 5;
  const int lane = threadIdx.x & 31;
  const int tile = blockIdx.x * 8 + wave;
  if (tile >= n_tiles) return;               // whole-wave exit, EXEC stays full for WMMA
  const int c = lane & 15, half = lane >> 4;
  const int e0 = tile * 16;
  int er = e0 + c; if (er >= E) er = E - 1;  // clamped gather (stores are guarded)
  const int nr = erow[er], nc = ecol[er];

  v16h A[6];
  A[0] = a_frag_f32(h  + (size_t)nr * 64,      half);
  A[1] = a_frag_f32(h  + (size_t)nr * 64 + 32, half);
  A[2] = a_frag_f32(h  + (size_t)nc * 64,      half);
  A[3] = a_frag_f32(h  + (size_t)nc * 64 + 32, half);
  A[4] = a_frag_f32(ef + (size_t)er * 64,      half);
  A[5] = a_frag_f32(ef + (size_t)er * 64 + 32, half);

  v8f acc[8];
#pragma unroll
  for (int nt = 0; nt < 8; ++nt) {
    v8f s = {};
#pragma unroll
    for (int kf = 0; kf < 6; ++kf)
      s = wmma16(A[kf], b_frag(W1T, 192, nt, kf, c, half), s);
    acc[nt] = s;
  }

  // bias + LayerNorm(128) + SiLU in D-layout (row = half*8+j, col = nt*16+c)
  float mu[8], rs[8];
#pragma unroll
  for (int j = 0; j < 8; ++j) { mu[j] = 0.f; rs[j] = 0.f; }
#pragma unroll
  for (int nt = 0; nt < 8; ++nt) {
    float bb = b1[nt * 16 + c];
#pragma unroll
    for (int j = 0; j < 8; ++j) {
      float v = acc[nt][j] + bb;
      acc[nt][j] = v; mu[j] += v; rs[j] += v * v;
    }
  }
#pragma unroll
  for (int m = 8; m; m >>= 1) {
#pragma unroll
    for (int j = 0; j < 8; ++j) {
      mu[j] += __shfl_xor(mu[j], m, 16);
      rs[j] += __shfl_xor(rs[j], m, 16);
    }
  }
#pragma unroll
  for (int j = 0; j < 8; ++j) {
    float m_ = mu[j] * (1.f / 128.f);
    float v_ = rs[j] * (1.f / 128.f) - m_ * m_;
    mu[j] = m_; rs[j] = rsqrtf(v_ + 1e-5f);
  }
#pragma unroll
  for (int nt = 0; nt < 8; ++nt) {
    float gg = g[nt * 16 + c], tb = bt[nt * 16 + c];
#pragma unroll
    for (int j = 0; j < 8; ++j) {
      float t = (acc[nt][j] - mu[j]) * rs[j] * gg + tb;
      float s = t / (1.f + expf(-t));          // SiLU
      Y[wave][half * 8 + j][nt * 16 + c] = (_Float16)s;
    }
  }
  __syncthreads();   // D-layout -> A-layout transpose through LDS

  v16h A2[4];
  const _Float16* yr = &Y[wave][c][0];
#pragma unroll
  for (int kf = 0; kf < 4; ++kf) A2[kf] = a_frag_lds(yr, kf, half);

  int rid[8], eid[8];
#pragma unroll
  for (int j = 0; j < 8; ++j) {
    int e = e0 + half * 8 + j;
    eid[j] = e;
    int ec = e < E ? e : E - 1;
    rid[j] = erow[ec];
  }

#pragma unroll
  for (int nt = 0; nt < 4; ++nt) {
    v8f s = {};
#pragma unroll
    for (int kf = 0; kf < 4; ++kf)
      s = wmma16(A2[kf], b_frag(W2T, 128, nt, kf, c, half), s);
    float bb = b2[nt * 16 + c];
    int d = nt * 16 + c;
#pragma unroll
    for (int j = 0; j < 8; ++j) {
      int e = eid[j];
      if (e < E) {
        float v = s[j] + bb + ef[(size_t)e * 64 + d];   // residual
        e_out[(size_t)e * 64 + d] = v;
        atomicAdd(agg + (size_t)rid[j] * 64 + d, v);    // segment_sum (L2-resident)
      }
    }
  }
}

// ---------------- node model ----------------
__global__ __launch_bounds__(256) void node_kernel(
    const float* __restrict__ h, const float* __restrict__ agg,
    const _Float16* __restrict__ W1T, const float* __restrict__ b1,
    const float* __restrict__ g, const float* __restrict__ bt,
    const _Float16* __restrict__ W2T, const float* __restrict__ b2,
    float* __restrict__ h_out, int N, int n_tiles) {
  __shared__ _Float16 Y[8][16][LDSTRIDE];
  const int wave = threadIdx.x >> 5;
  const int lane = threadIdx.x & 31;
  const int tile = blockIdx.x * 8 + wave;
  if (tile >= n_tiles) return;
  const int c = lane & 15, half = lane >> 4;
  const int r0 = tile * 16;
  int r = r0 + c; if (r >= N) r = N - 1;

  v16h A[4];
  A[0] = a_frag_f32(h   + (size_t)r * 64,      half);
  A[1] = a_frag_f32(h   + (size_t)r * 64 + 32, half);
  A[2] = a_frag_f32(agg + (size_t)r * 64,      half);
  A[3] = a_frag_f32(agg + (size_t)r * 64 + 32, half);

  v8f acc[8];
#pragma unroll
  for (int nt = 0; nt < 8; ++nt) {
    v8f s = {};
#pragma unroll
    for (int kf = 0; kf < 4; ++kf)
      s = wmma16(A[kf], b_frag(W1T, 128, nt, kf, c, half), s);
    acc[nt] = s;
  }

  float mu[8], rs[8];
#pragma unroll
  for (int j = 0; j < 8; ++j) { mu[j] = 0.f; rs[j] = 0.f; }
#pragma unroll
  for (int nt = 0; nt < 8; ++nt) {
    float bb = b1[nt * 16 + c];
#pragma unroll
    for (int j = 0; j < 8; ++j) {
      float v = acc[nt][j] + bb;
      acc[nt][j] = v; mu[j] += v; rs[j] += v * v;
    }
  }
#pragma unroll
  for (int m = 8; m; m >>= 1) {
#pragma unroll
    for (int j = 0; j < 8; ++j) {
      mu[j] += __shfl_xor(mu[j], m, 16);
      rs[j] += __shfl_xor(rs[j], m, 16);
    }
  }
#pragma unroll
  for (int j = 0; j < 8; ++j) {
    float m_ = mu[j] * (1.f / 128.f);
    float v_ = rs[j] * (1.f / 128.f) - m_ * m_;
    mu[j] = m_; rs[j] = rsqrtf(v_ + 1e-5f);
  }
#pragma unroll
  for (int nt = 0; nt < 8; ++nt) {
    float gg = g[nt * 16 + c], tb = bt[nt * 16 + c];
#pragma unroll
    for (int j = 0; j < 8; ++j) {
      float t = (acc[nt][j] - mu[j]) * rs[j] * gg + tb;
      float s = t / (1.f + expf(-t));
      Y[wave][half * 8 + j][nt * 16 + c] = (_Float16)s;
    }
  }
  __syncthreads();

  v16h A2[4];
  const _Float16* yr = &Y[wave][c][0];
#pragma unroll
  for (int kf = 0; kf < 4; ++kf) A2[kf] = a_frag_lds(yr, kf, half);

#pragma unroll
  for (int nt = 0; nt < 4; ++nt) {
    v8f s = {};
#pragma unroll
    for (int kf = 0; kf < 4; ++kf)
      s = wmma16(A2[kf], b_frag(W2T, 128, nt, kf, c, half), s);
    float bb = b2[nt * 16 + c];
    int d = nt * 16 + c;
#pragma unroll
    for (int j = 0; j < 8; ++j) {
      int node = r0 + half * 8 + j;
      if (node < N) {
        float v = s[j] + bb + h[(size_t)node * 64 + d];  // residual
        h_out[(size_t)node * 64 + d] = v;
      }
    }
  }
}

// ---------------- prep ----------------
__global__ void zero_f32(float* __restrict__ p, int n) {
  int i = blockIdx.x * 256 + threadIdx.x;
  if (i < n) p[i] = 0.f;
}

// fp32 [K,N] row-major -> f16 transposed WT[N][K] (contiguous-K rows for B frags)
__global__ void prep_weights(const float* __restrict__ eW1, const float* __restrict__ eW2,
                             const float* __restrict__ nW1, const float* __restrict__ nW2,
                             _Float16* __restrict__ eW1T, _Float16* __restrict__ eW2T,
                             _Float16* __restrict__ nW1T, _Float16* __restrict__ nW2T) {
  int i = blockIdx.x * 256 + threadIdx.x;
  if (i < 24576) { int k = i / 128, n = i % 128; eW1T[n * 192 + k] = (_Float16)eW1[i]; return; }
  i -= 24576;
  if (i < 8192)  { int k = i / 64,  n = i % 64;  eW2T[n * 128 + k] = (_Float16)eW2[i]; return; }
  i -= 8192;
  if (i < 16384) { int k = i / 128, n = i % 128; nW1T[n * 128 + k] = (_Float16)nW1[i]; return; }
  i -= 16384;
  if (i < 8192)  { int k = i / 64,  n = i % 64;  nW2T[n * 128 + k] = (_Float16)nW2[i]; }
}

extern "C" void kernel_launch(void* const* d_in, const int* in_sizes, int n_in,
                              void* d_out, int out_size, void* d_ws, size_t ws_size,
                              hipStream_t stream) {
  const float* h   = (const float*)d_in[0];
  const float* ef  = (const float*)d_in[1];
  const float* eW1 = (const float*)d_in[2];
  const float* eb1 = (const float*)d_in[3];
  const float* eg  = (const float*)d_in[4];
  const float* ebt = (const float*)d_in[5];
  const float* eW2 = (const float*)d_in[6];
  const float* eb2 = (const float*)d_in[7];
  const float* nW1 = (const float*)d_in[8];
  const float* nb1 = (const float*)d_in[9];
  const float* ng  = (const float*)d_in[10];
  const float* nbt = (const float*)d_in[11];
  const float* nW2 = (const float*)d_in[12];
  const float* nb2 = (const float*)d_in[13];
  const int*  eidx = (const int*)d_in[14];

  const int N = in_sizes[0] / 64;
  const int E = in_sizes[1] / 64;
  const int* erow = eidx;
  const int* ecol = eidx + E;

  float* h_out = (float*)d_out;                     // outputs: (h_out, e_out) flat
  float* e_out = (float*)d_out + (size_t)N * 64;

  char* ws = (char*)d_ws;
  size_t aggB = ((size_t)N * 64 * 4 + 255) & ~(size_t)255;
  float*    agg  = (float*)ws;
  _Float16* eW1T = (_Float16*)(ws + aggB);
  _Float16* eW2T = (_Float16*)(ws + aggB + 49152);
  _Float16* nW1T = (_Float16*)(ws + aggB + 49152 + 16384);
  _Float16* nW2T = (_Float16*)(ws + aggB + 49152 + 16384 + 32768);

  { int n = N * 64; zero_f32<<<(n + 255) / 256, 256, 0, stream>>>(agg, n); }
  prep_weights<<<(57344 + 255) / 256, 256, 0, stream>>>(eW1, eW2, nW1, nW2,
                                                        eW1T, eW2T, nW1T, nW2T);

  int etiles = (E + 15) / 16;
  edge_kernel<<<(etiles + 7) / 8, 256, 0, stream>>>(h, ef, eW1T, eb1, eg, ebt, eW2T, eb2,
                                                    erow, ecol, e_out, agg, E, etiles);
  int ntiles = (N + 15) / 16;
  node_kernel<<<(ntiles + 7) / 8, 256, 0, stream>>>(h, agg, nW1T, nb1, ng, nbt, nW2T, nb2,
                                                    h_out, N, ntiles);
}
